// Decoder_22557168238911
// MI455X (gfx1250) — compile-verified, matrix operand order checked by reading, not run on hardware
//
#include <hip/hip_runtime.h>
#include <hip/hip_bf16.h>

typedef __attribute__((ext_vector_type(16))) __bf16 v16bf;
typedef __attribute__((ext_vector_type(8)))  float  v8f;
typedef unsigned short u16;

// ---------------- helpers ----------------
__device__ __forceinline__ u16 f2bf(float f) {
  unsigned int u = __float_as_uint(f);
  unsigned int r = u + 0x7FFFu + ((u >> 16) & 1u);   // round-to-nearest-even
  return (u16)(r >> 16);
}
__device__ __forceinline__ float sigf(float x) { return 1.0f / (1.0f + __expf(-x)); }

// ---------------- generic bf16 fragment packer ----------------
// Packs up to 3 concatenated row-major fp32 segments [s0|s1|s2] (widths K0,K1,K2)
// into WMMA 16x32 bf16 fragment order. tile = rt*KT + kt; 32 lanes x 16 halves.
// Zero-fills K beyond K0+K1+K2 (K padding to multiples of 32).
__global__ void pack_tiles(const float* __restrict__ s0, int K0,
                           const float* __restrict__ s1, int K1,
                           const float* __restrict__ s2, int K2,
                           int KT, u16* __restrict__ dst) {
  int tile = blockIdx.x;
  int kt   = tile % KT;
  int lane = threadIdx.x;
  int row  = (tile / KT) * 16 + (lane & 15);
  int half = lane >> 4;
  u16* d = dst + (size_t)tile * 512 + lane * 16;
  for (int h = 0; h < 16; ++h) {
    int i = h >> 1, p = h & 1;
    int kin = (i < 4) ? (2 * i) : (16 + 2 * (i - 4));
    int k = kt * 32 + kin + 8 * half + p;
    float v = 0.0f;
    if (k < K0)                    v = s0[(size_t)row * K0 + k];
    else if (k < K0 + K1)          v = s1[(size_t)row * K1 + (k - K0)];
    else if (k < K0 + K1 + K2)     v = s2[(size_t)row * K2 + (k - K0 - K1)];
    d[h] = f2bf(v);
  }
}

// ---- WMMA GEMM, 2 m-tiles per wave (B-fragment reused for 2 WMMAs) ----
// MP = MT/2 m-tile pairs. C[M x N] = Apack * Bpack^T (+bias) (opt ReLU).
__global__ void gemm2_bf16_wmma(const u16* __restrict__ Ap, const u16* __restrict__ Bp,
                                const float* __restrict__ bias, float* __restrict__ C,
                                int MP, int NT, int KT, int ldc, int relu) {
  int wave = (int)((blockIdx.x * blockDim.x + threadIdx.x) >> 5);
  int lane = threadIdx.x & 31;
  if (wave >= MP * NT) return;
  int mp = wave % MP;
  int nt = wave / MP;
  const u16* a0 = Ap + (size_t)(2 * mp) * KT * 512 + lane * 16;
  const u16* a1 = a0 + (size_t)KT * 512;
  const u16* bp = Bp + (size_t)nt * KT * 512 + lane * 16;
  v8f acc0 = {}, acc1 = {};
  for (int kt = 0; kt < KT; ++kt) {
    __builtin_prefetch(bp + 1024, 0, 1);
    v16bf b  = *(const v16bf*)bp;
    v16bf x0 = *(const v16bf*)a0;
    v16bf x1 = *(const v16bf*)a1;
    acc0 = __builtin_amdgcn_wmma_f32_16x16x32_bf16(false, x0, false, b,
                                                   (short)0, acc0, false, false);
    acc1 = __builtin_amdgcn_wmma_f32_16x16x32_bf16(false, x1, false, b,
                                                   (short)0, acc1, false, false);
    a0 += 512; a1 += 512; bp += 512;
  }
  int n  = nt * 16 + (lane & 15);
  int m0 = 2 * mp * 16 + ((lane >> 4) << 3);
  float bv = bias ? bias[n] : 0.0f;
  #pragma unroll
  for (int r = 0; r < 8; ++r) {
    float v0 = acc0[r] + bv;
    float v1 = acc1[r] + bv;
    if (relu) { v0 = fmaxf(v0, 0.0f); v1 = fmaxf(v1, 0.0f); }
    C[(size_t)(m0 + r) * ldc + n]      = v0;
    C[(size_t)(m0 + 16 + r) * ldc + n] = v1;
  }
}

// ---------------- small utility kernels ----------------
__global__ void zero_f32(float* p, int n) {
  int i = blockIdx.x * blockDim.x + threadIdx.x;
  if (i < n) p[i] = 0.0f;
}
__global__ void sum_bias(const float* a, const float* b, float* o, int n) {
  int i = blockIdx.x * blockDim.x + threadIdx.x;
  if (i < n) o[i] = a[i] + b[i];
}
// dec_in[t][b][m] = (t==0) ? 0 : decoder_inputs[b][m][t-1]   (B=32,NMEL=80,T=300)
__global__ void build_decin(const float* __restrict__ di, float* __restrict__ o) {
  int idx = blockIdx.x * blockDim.x + threadIdx.x;
  if (idx >= 300 * 32 * 80) return;
  int m = idx % 80, b = (idx / 80) % 32, t = idx / (80 * 32);
  o[idx] = (t == 0) ? 0.0f : di[((size_t)b * 80 + m) * 300 + (t - 1)];
}
// LSTM gate math: z layout [i|f|g|o] each H wide; B=32
__global__ void lstm_gates(const float* __restrict__ z, float* __restrict__ h,
                           float* __restrict__ c, int H) {
  int idx = blockIdx.x * blockDim.x + threadIdx.x;
  if (idx >= 32 * H) return;
  int b = idx / H, u = idx % H;
  const float* zr = z + (size_t)b * 4 * H;
  float si = sigf(zr[u]);
  float sf = sigf(zr[H + u]);
  float g  = tanhf(zr[2 * H + u]);
  float so = sigf(zr[3 * H + u]);
  float cn = sf * c[idx] + si * g;
  c[idx] = cn;
  h[idx] = so * tanhf(cn);
}
// e[b,t] = v . tanh(q[b] + W_loc @ conv([aw;awc])[b,:,t] + pm[b,t])
// One block handles 8 consecutive t (W_loc/q cached per thread).
__global__ void attn_energy8(const float* __restrict__ q, const float* __restrict__ pm,
                             const float* __restrict__ aw, const float* __restrict__ awc,
                             const float* __restrict__ conv, const float* __restrict__ Wloc,
                             const float* __restrict__ vv, float* __restrict__ e) {
  int b  = blockIdx.x / 25;
  int t0 = (blockIdx.x % 25) * 8;
  int a  = threadIdx.x;                       // 128 threads
  __shared__ float loc[8][32];
  __shared__ float red[8][4];
  if (a < 32) {
    const float* w0 = conv + a * 62;          // (NF,2,31)
    for (int j = 0; j < 8; ++j) {
      int t = t0 + j;
      float acc = 0.0f;
      for (int k = 0; k < 31; ++k) {
        int tt = t + k - 15;
        if (tt >= 0 && tt < 200) {
          acc += aw [b * 200 + tt] * w0[k];
          acc += awc[b * 200 + tt] * w0[31 + k];
        }
      }
      loc[j][a] = acc;
    }
  }
  __syncthreads();
  float wl[32];
  #pragma unroll
  for (int f = 0; f < 32; ++f) wl[f] = Wloc[a * 32 + f];
  float qa = q[b * 128 + a];
  float va = vv[a];
  for (int j = 0; j < 8; ++j) {
    int t = t0 + j;
    float s = qa + pm[((size_t)b * 200 + t) * 128 + a];
    #pragma unroll
    for (int f = 0; f < 32; ++f) s += wl[f] * loc[j][f];
    float contrib = va * tanhf(s);
    for (int o = 16; o > 0; o >>= 1) contrib += __shfl_down(contrib, o, 32);
    if ((a & 31) == 0) red[j][a >> 5] = contrib;
  }
  __syncthreads();
  if (a < 8) e[b * 200 + t0 + a] = red[a][0] + red[a][1] + red[a][2] + red[a][3];
}
// masked softmax over T_in, context, cumulative update, writes alignment output
__global__ void softmax_ctx(const float* __restrict__ e, const int* __restrict__ mlen,
                            const float* __restrict__ memory, float* __restrict__ aw,
                            float* __restrict__ awc, float* __restrict__ ctx,
                            float* __restrict__ align_out, int t_step) {
  int b = blockIdx.x, tid = threadIdx.x;       // 256 threads
  __shared__ float w[256];
  __shared__ float red[256];
  int L = mlen[b];
  float val = (tid < 200 && tid < L) ? e[b * 200 + tid] : -3.0e38f;
  red[tid] = val; __syncthreads();
  for (int s = 128; s > 0; s >>= 1) { if (tid < s) red[tid] = fmaxf(red[tid], red[tid + s]); __syncthreads(); }
  float mx = red[0]; __syncthreads();
  float ex = (tid < 200 && tid < L) ? __expf(val - mx) : 0.0f;
  red[tid] = ex; __syncthreads();
  for (int s = 128; s > 0; s >>= 1) { if (tid < s) red[tid] += red[tid + s]; __syncthreads(); }
  float inv = 1.0f / red[0];
  __syncthreads();
  float wn = ex * inv;
  w[tid] = wn;
  if (tid < 200) {
    aw[b * 200 + tid]  = wn;
    awc[b * 200 + tid] += wn;
    align_out[((size_t)b * 300 + t_step) * 200 + tid] = wn;
  }
  __syncthreads();
  for (int eo = tid; eo < 512; eo += 256) {
    float acc = 0.0f;
    for (int tt = 0; tt < 200; ++tt)
      acc += w[tt] * memory[((size_t)b * 200 + tt) * 512 + eo];
    ctx[b * 512 + eo] = acc;
  }
}
// mel scatter (transposed layout) + gate dot from [dec_h|ctx]
__global__ void out_write(const float* __restrict__ melbuf, const float* __restrict__ dh,
                          const float* __restrict__ ctx, const float* __restrict__ Wg,
                          const float* __restrict__ bg, float* __restrict__ out, int t) {
  int b = blockIdx.x, tid = threadIdx.x;       // 128 threads
  if (tid < 80) out[((size_t)b * 80 + tid) * 300 + t] = melbuf[b * 80 + tid];
  float acc = 0.0f;
  for (int k = tid; k < 1536; k += 128) {
    float hv = (k < 1024) ? dh[b * 1024 + k] : ctx[b * 512 + (k - 1024)];
    acc += hv * Wg[k];
  }
  __shared__ float red[128];
  red[tid] = acc; __syncthreads();
  for (int s = 64; s > 0; s >>= 1) { if (tid < s) red[tid] += red[tid + s]; __syncthreads(); }
  if (tid == 0) out[768000 + (size_t)b * 300 + t] = red[0] + bg[0];
}

// ---------------- workspace layout (bytes) ----------------
static const size_t WA_PACK   = 0;                         // 4096x1792 u16
static const size_t WD_PACK   = 14680064;                  // 4096x2560 u16
static const size_t WMEM_PACK = 35651584;                  // 128x512 u16
static const size_t WPRE1_PK  = 35782656;                  // 256x96 u16
static const size_t WPRE2_PK  = 35831808;                  // 256x256 u16
static const size_t MEMA_PACK = 35962880;                  // 6400x512 u16
static const size_t PM_OFF    = 42516480;                  // 6400x128 f32
static const size_t DECIN_OFF = 45793280;                  // 9600x80 f32
static const size_t A1_PACK   = 48865280;                  // 9600x96 u16
static const size_t Z1_OFF    = 50708480;                  // 9600x256 f32
static const size_t A2_PACK   = 60538880;                  // 9600x256 u16
static const size_t PRE_OFF   = 65454080;                  // 9600x256 f32
static const size_t XA_PACK   = 75284480;                  // 32x1792 u16
static const size_t ZA_OFF    = 75399168;                  // 32x4096 f32
static const size_t XD_PACK   = 75923456;                  // 32x2560 u16
static const size_t ZD_OFF    = 76087296;                  // 32x4096 f32
static const size_t BA_OFF    = 76611584;                  // 4096 f32
static const size_t BD_OFF    = 76627968;                  // 4096 f32
static const size_t Q_OFF     = 76644352;                  // 32x128 f32
static const size_t E_OFF     = 76660736;                  // 32x200 f32
static const size_t ST_OFF    = 76686336;                  // states: 160256 f32
static const size_t WQ_PACK   = 77327360;                  // 128x1024 u16
static const size_t WPROJ_PK  = 77589504;                  // 80x1536 u16
static const size_t QA_PACK   = 77835264;                  // 32x1024 u16
static const size_t HC_PACK   = 77966336;                  // 32x1536 u16
static const size_t MELB_OFF  = 78064640;                  // 32x80 f32

extern "C" void kernel_launch(void* const* d_in, const int* in_sizes, int n_in,
                              void* d_out, int out_size, void* d_ws, size_t ws_size,
                              hipStream_t stream) {
  const float* memory   = (const float*)d_in[0];
  const float* dec_inp  = (const float*)d_in[1];
  const int*   mlen     = (const int*)  d_in[2];
  const float* W_pre1   = (const float*)d_in[3];
  const float* W_pre2   = (const float*)d_in[4];
  const float* W_ih_a   = (const float*)d_in[5];
  const float* W_hh_a   = (const float*)d_in[6];
  const float* b_ih_a   = (const float*)d_in[7];
  const float* b_hh_a   = (const float*)d_in[8];
  const float* W_query  = (const float*)d_in[9];
  const float* W_mem    = (const float*)d_in[10];
  const float* vvec     = (const float*)d_in[11];
  const float* conv_loc = (const float*)d_in[12];
  const float* W_loc    = (const float*)d_in[13];
  const float* W_ih_d   = (const float*)d_in[14];
  const float* W_hh_d   = (const float*)d_in[15];
  const float* b_ih_d   = (const float*)d_in[16];
  const float* b_hh_d   = (const float*)d_in[17];
  const float* W_proj   = (const float*)d_in[18];
  const float* b_proj   = (const float*)d_in[19];
  const float* W_gate   = (const float*)d_in[20];
  const float* b_gate   = (const float*)d_in[21];

  char* ws = (char*)d_ws;
  u16*  wa   = (u16*)(ws + WA_PACK);
  u16*  wd   = (u16*)(ws + WD_PACK);
  u16*  wmem = (u16*)(ws + WMEM_PACK);
  u16*  wp1  = (u16*)(ws + WPRE1_PK);
  u16*  wp2  = (u16*)(ws + WPRE2_PK);
  u16*  mema = (u16*)(ws + MEMA_PACK);
  float* pm    = (float*)(ws + PM_OFF);
  float* decin = (float*)(ws + DECIN_OFF);
  u16*  a1p  = (u16*)(ws + A1_PACK);
  float* z1   = (float*)(ws + Z1_OFF);
  u16*  a2p  = (u16*)(ws + A2_PACK);
  float* pre  = (float*)(ws + PRE_OFF);
  u16*  xap  = (u16*)(ws + XA_PACK);
  float* za   = (float*)(ws + ZA_OFF);
  u16*  xdp  = (u16*)(ws + XD_PACK);
  float* zd   = (float*)(ws + ZD_OFF);
  float* ba   = (float*)(ws + BA_OFF);
  float* bd   = (float*)(ws + BD_OFF);
  float* q    = (float*)(ws + Q_OFF);
  float* ebuf = (float*)(ws + E_OFF);
  float* st   = (float*)(ws + ST_OFF);
  u16*  wq   = (u16*)(ws + WQ_PACK);
  u16*  wpp  = (u16*)(ws + WPROJ_PK);
  u16*  qap  = (u16*)(ws + QA_PACK);
  u16*  hcp  = (u16*)(ws + HC_PACK);
  float* melb = (float*)(ws + MELB_OFF);
  float* attn_h = st;              // 32x1024
  float* attn_c = st + 32768;
  float* dec_h  = st + 65536;
  float* dec_c  = st + 98304;
  float* aw     = st + 131072;     // 32x200
  float* awc    = st + 137472;
  float* ctx    = st + 143872;     // 32x512
  float* out    = (float*)d_out;
  float* align_out = out + 777600; // mel:768000 gate:9600 align:1920000

  const float* nul = nullptr;

  // ---- setup: pack weights (bf16 fragment order), combined biases ----
  hipLaunchKernelGGL(pack_tiles, dim3(256 * 56), dim3(32), 0, stream,
                     W_ih_a, 768, W_hh_a, 1024, nul, 0, 56, wa);
  hipLaunchKernelGGL(pack_tiles, dim3(256 * 80), dim3(32), 0, stream,
                     W_ih_d, 1536, W_hh_d, 1024, nul, 0, 80, wd);
  hipLaunchKernelGGL(pack_tiles, dim3(8 * 16), dim3(32), 0, stream,
                     W_mem, 512, nul, 0, nul, 0, 16, wmem);
  hipLaunchKernelGGL(pack_tiles, dim3(16 * 3), dim3(32), 0, stream,
                     W_pre1, 80, nul, 0, nul, 0, 3, wp1);
  hipLaunchKernelGGL(pack_tiles, dim3(16 * 8), dim3(32), 0, stream,
                     W_pre2, 256, nul, 0, nul, 0, 8, wp2);
  hipLaunchKernelGGL(pack_tiles, dim3(8 * 32), dim3(32), 0, stream,
                     W_query, 1024, nul, 0, nul, 0, 32, wq);
  hipLaunchKernelGGL(pack_tiles, dim3(5 * 48), dim3(32), 0, stream,
                     W_proj, 1536, nul, 0, nul, 0, 48, wpp);
  hipLaunchKernelGGL(sum_bias, dim3(16), dim3(256), 0, stream, b_ih_a, b_hh_a, ba, 4096);
  hipLaunchKernelGGL(sum_bias, dim3(16), dim3(256), 0, stream, b_ih_d, b_hh_d, bd, 4096);

  // ---- processed_memory = memory @ W_mem^T (6400x128) via WMMA ----
  hipLaunchKernelGGL(pack_tiles, dim3(400 * 16), dim3(32), 0, stream,
                     memory, 512, nul, 0, nul, 0, 16, mema);
  hipLaunchKernelGGL(gemm2_bf16_wmma, dim3(200), dim3(256), 0, stream,
                     mema, wmem, nul, pm, 200, 8, 16, 128, 0);

  // ---- prenet over all (t,b): relu(relu(decin@W1^T)@W2^T) via WMMA ----
  hipLaunchKernelGGL(build_decin, dim3(3000), dim3(256), 0, stream, dec_inp, decin);
  hipLaunchKernelGGL(pack_tiles, dim3(600 * 3), dim3(32), 0, stream,
                     decin, 80, nul, 0, nul, 0, 3, a1p);
  hipLaunchKernelGGL(gemm2_bf16_wmma, dim3(600), dim3(256), 0, stream,
                     a1p, wp1, nul, z1, 300, 16, 3, 256, 1);
  hipLaunchKernelGGL(pack_tiles, dim3(600 * 8), dim3(32), 0, stream,
                     z1, 256, nul, 0, nul, 0, 8, a2p);
  hipLaunchKernelGGL(gemm2_bf16_wmma, dim3(600), dim3(256), 0, stream,
                     a2p, wp2, nul, pre, 300, 16, 8, 256, 1);

  // ---- zero recurrent state ----
  hipLaunchKernelGGL(zero_f32, dim3(627), dim3(256), 0, stream, st, 160256);

  // ---- autoregressive loop ----
  for (int t = 0; t < 300; ++t) {
    // attention LSTM: x = [pre_t | ctx | attn_h] (1792)
    hipLaunchKernelGGL(pack_tiles, dim3(2 * 56), dim3(32), 0, stream,
                       pre + (size_t)t * 32 * 256, 256, ctx, 512, attn_h, 1024, 56, xap);
    hipLaunchKernelGGL(gemm2_bf16_wmma, dim3(32), dim3(256), 0, stream,
                       xap, wa, ba, za, 1, 256, 56, 4096, 0);
    hipLaunchKernelGGL(lstm_gates, dim3(128), dim3(256), 0, stream, za, attn_h, attn_c, 1024);
    // attention: q = attn_h @ W_query^T via WMMA
    hipLaunchKernelGGL(pack_tiles, dim3(2 * 32), dim3(32), 0, stream,
                       attn_h, 1024, nul, 0, nul, 0, 32, qap);
    hipLaunchKernelGGL(gemm2_bf16_wmma, dim3(1), dim3(256), 0, stream,
                       qap, wq, nul, q, 1, 8, 32, 128, 0);
    hipLaunchKernelGGL(attn_energy8, dim3(800), dim3(128), 0, stream,
                       q, pm, aw, awc, conv_loc, W_loc, vvec, ebuf);
    hipLaunchKernelGGL(softmax_ctx, dim3(32), dim3(256), 0, stream,
                       ebuf, mlen, memory, aw, awc, ctx, align_out, t);
    // decoder LSTM: x = [attn_h | ctx | dec_h] (2560)
    hipLaunchKernelGGL(pack_tiles, dim3(2 * 80), dim3(32), 0, stream,
                       attn_h, 1024, ctx, 512, dec_h, 1024, 80, xdp);
    hipLaunchKernelGGL(gemm2_bf16_wmma, dim3(32), dim3(256), 0, stream,
                       xdp, wd, bd, zd, 1, 256, 80, 4096, 0);
    hipLaunchKernelGGL(lstm_gates, dim3(128), dim3(256), 0, stream, zd, dec_h, dec_c, 1024);
    // outputs: mel = [dec_h|ctx] @ W_proj^T + b_proj via WMMA; gate via reduction
    hipLaunchKernelGGL(pack_tiles, dim3(2 * 48), dim3(32), 0, stream,
                       dec_h, 1024, ctx, 512, nul, 0, 48, hcp);
    hipLaunchKernelGGL(gemm2_bf16_wmma, dim3(1), dim3(256), 0, stream,
                       hcp, wpp, b_proj, melb, 1, 5, 48, 80, 0);
    hipLaunchKernelGGL(out_write, dim3(32), dim3(128), 0, stream,
                       melb, dec_h, ctx, W_gate, b_gate, out, t);
  }
}
